// CPKAN_21569325761073
// MI455X (gfx1250) — matrix operand order repaired and checked by reading, not compiled, not run on hardware
//
#include <hip/hip_runtime.h>
#include <hip/hip_bf16.h>
#include <math.h>

#define DD 64
#define TT 64
#define BB 4096
#define NLAYER 2

typedef __attribute__((ext_vector_type(16))) _Float16 v16h;
typedef __attribute__((ext_vector_type(8)))  _Float16 v8h;
typedef __attribute__((ext_vector_type(8)))  float    v8f;

// ---------------------------------------------------------------------------
// WMMA helpers (CDNA5 / gfx1250, wave32)
// ---------------------------------------------------------------------------
__device__ __forceinline__ v8f wmma_f16(v16h a, v16h b, v8f c) {
    // D = A(16x32 f16) * B(32x16 f16) + C(16x16 f32)
    return __builtin_amdgcn_wmma_f32_16x16x32_f16(
        /*neg_a=*/false, a, /*neg_b=*/false, b,
        /*c_mod=*/(short)0, c, /*reuse_a=*/false, /*reuse_b=*/false);
}

// A-matrix fragment: 16x32 f16 tile (row-major in LDS, row stride `ld` halfs).
// ISA 7.12.2: lanes 0-15 hold row M=lane, K={0..7,16..23}; lanes 16-31 hold
// row M=lane-16, K={8..15,24..31}.  Two aligned 16B LDS loads per lane.
__device__ __forceinline__ v16h load_a_frag(const _Float16* tile, int ld, int lane) {
    int row = lane & 15;
    int kb  = (lane >> 4) << 3;            // 0 or 8
    const _Float16* p = tile + row * ld;
    v8h lo = *(const v8h*)(p + kb);        // K = kb .. kb+7
    v8h hi = *(const v8h*)(p + 16 + kb);   // K = 16+kb .. 16+kb+7
    v16h a;
#pragma unroll
    for (int e = 0; e < 8; ++e) { a[e] = lo[e]; a[8 + e] = hi[e]; }
    return a;
}

// B-matrix fragment from PACKED weights: fragment-major layout, each fragment
// is 32 lanes x 16 halfs contiguous -> two coalesced global_load_b128 per lane.
__device__ __forceinline__ v16h load_b_frag_packed(const _Float16* p, int lane) {
    const v8h* q = (const v8h*)(p + lane * 16);
    v8h lo = q[0];
    v8h hi = q[1];
    v16h b;
#pragma unroll
    for (int e = 0; e < 8; ++e) { b[e] = lo[e]; b[8 + e] = hi[e]; }
    return b;
}

// Block-level GEMM: Y(f32, MT*16 x NT*16) = act( X(f16) @ W(f16,packed) + bias ).
// X in LDS, W packed fragment-major in global (L2-resident), Y in LDS.
// Fully unrolled: MT*NT must be a multiple of NWAVES (1-2 tiles per wave).
// ACT: 0=none, 1=leaky-relu(0.01), 2=relu
template <int KT, int ACT, int MT, int NT, int NWAVES>
__device__ __forceinline__ void gemm_block(const _Float16* X, int ldx,
                                           const _Float16* Wp,
                                           float* Y, int ldy,
                                           const float* bias,
                                           int wave, int lane) {
#pragma unroll
    for (int i = 0; i < (MT * NT) / NWAVES; ++i) {
        int t  = wave + i * NWAVES;
        int mt = t / NT, nt = t % NT;
        v16h afr[KT];
#pragma unroll
        for (int kt = 0; kt < KT; ++kt)
            afr[kt] = load_a_frag(X + (mt * 16) * ldx + kt * 32, ldx, lane);
        v8f c = {};
#pragma unroll
        for (int kt = 0; kt < KT; ++kt) {
            v16h b = load_b_frag_packed(Wp + (size_t)(kt * NT + nt) * 512, lane);
            c = wmma_f16(afr[kt], b, c);
        }
        // C/D layout: VGPR i -> row r0+i, col = nt*16 + (lane&15)
        int col = nt * 16 + (lane & 15);
        int r0  = mt * 16 + ((lane >> 4) << 3);
        float bv = bias ? bias[col] : 0.0f;
#pragma unroll
        for (int j = 0; j < 8; ++j) {
            float v = c[j] + bv;
            if (ACT == 1) v = v > 0.0f ? v : 0.01f * v;
            if (ACT == 2) v = v > 0.0f ? v : 0.0f;
            Y[(r0 + j) * ldy + col] = v;
        }
    }
}

__device__ __forceinline__ float sigmoidf_(float x) { return 1.0f / (1.0f + __expf(-x)); }

// Row-dot with a 64-long f16 vector, vector loads (8x global_load_b128)
__device__ __forceinline__ float rowdot64(const float* yrow, const _Float16* w) {
    const v8h* wq = (const v8h*)w;
    float acc = 0.0f;
#pragma unroll
    for (int g = 0; g < 8; ++g) {
        v8h wv = wq[g];
#pragma unroll
        for (int e = 0; e < 8; ++e) acc += yrow[g * 8 + e] * (float)wv[e];
    }
    return acc;
}

// ---------------------------------------------------------------------------
// Kernel: f32 -> f16 plain conversion (vectors: ta_w2, ea_w3)
// ---------------------------------------------------------------------------
__global__ void cvt_f16_kernel(const float* __restrict__ src,
                               _Float16* __restrict__ dst, int n) {
    int i = blockIdx.x * blockDim.x + threadIdx.x;
    if (i < n) dst[i] = (_Float16)src[i];
}

// ---------------------------------------------------------------------------
// Kernel: pack f32 weight [K][N] into f16 WMMA B-fragment-major layout.
// Fragment id = kt*NT + nt ; inside a fragment: lane L (0..31) x element e:
//   K = kt*32 + ((e>>3)<<4) + ((L>>4)<<3) + (e&7) ,  N = nt*16 + (L&15)
// ---------------------------------------------------------------------------
__global__ void pack_b_kernel(const float* __restrict__ src,
                              _Float16* __restrict__ dst, int K, int N) {
    int i = blockIdx.x * blockDim.x + threadIdx.x;
    if (i >= K * N) return;
    int NT   = N >> 4;
    int frag = i >> 9;          // / 512
    int r    = i & 511;
    int lane = r >> 4, e = r & 15;
    int kt = frag / NT, nt = frag % NT;
    int k = kt * 32 + ((e >> 3) << 4) + ((lane >> 4) << 3) + (e & 7);
    int n = nt * 16 + (lane & 15);
    dst[i] = (_Float16)src[k * N + n];
}

// ---------------------------------------------------------------------------
// Kernel: knowledge attention for one (h,r,t) layer.  One block per batch b.
//   x = [h*r, t] (T x 128); y = lrelu(x@W1+b1); s = sigmoid(y@w2+b2);
//   a = softmax_T(s); out[b] = sum_t a[t]*t_emb[t]
// ---------------------------------------------------------------------------
__global__ __launch_bounds__(256) void ka_kernel(
    const int* __restrict__ h_idx, const int* __restrict__ r_idx,
    const int* __restrict__ t_idx,
    const float* __restrict__ entity, const float* __restrict__ relation,
    const _Float16* __restrict__ w1p, const float* __restrict__ b1,
    const _Float16* __restrict__ w2h, const float* __restrict__ b2,
    float* __restrict__ out) {
    __shared__ _Float16 X[TT][128];
    __shared__ _Float16 Te[TT][DD];
    __shared__ float    Y[TT][DD];
    __shared__ float    Srow[TT];
    __shared__ float    invsum;

    int b = blockIdx.x, tid = threadIdx.x;
    int wave = tid >> 5, lane = tid & 31;

    if (tid == 0) __builtin_prefetch(w1p, 0, 1);   // global_prefetch_b8

    // gather + build x : 4 threads per row, 16 cols each
    {
        int row = tid >> 2, q = tid & 3;
        int hi = h_idx[b * TT + row];
        int ri = r_idx[b * TT + row];
        int ti = t_idx[b * TT + row];
        const float* hp = entity + (size_t)hi * DD;
        const float* rp = relation + (size_t)ri * DD;
        const float* tp = entity + (size_t)ti * DD;
#pragma unroll
        for (int e = 0; e < 16; ++e) {
            int c = q * 16 + e;
            float tv = tp[c];
            X[row][c]      = (_Float16)(hp[c] * rp[c]);
            X[row][64 + c] = (_Float16)tv;
            Te[row][c]     = (_Float16)tv;
        }
    }
    __syncthreads();

    // y = lrelu(x @ W1 + b1): M=64,N=64,K=128
    gemm_block<4, 1, 4, 4, 8>(&X[0][0], 128, w1p, &Y[0][0], DD, b1, wave, lane);
    __syncthreads();

    // s = sigmoid(y @ w2 + b2) per row (w2 fetched as 8x b128)
    if (tid < TT) Srow[tid] = sigmoidf_(rowdot64(&Y[tid][0], w2h) + b2[0]);
    __syncthreads();

    // softmax over T
    if (tid == 0) {
        float m = -1e30f;
        for (int t = 0; t < TT; ++t) m = fmaxf(m, Srow[t]);
        float s = 0.0f;
        for (int t = 0; t < TT; ++t) { float e = __expf(Srow[t] - m); Srow[t] = e; s += e; }
        invsum = 1.0f / s;
    }
    __syncthreads();

    // out[b][d] = invsum * sum_t Srow[t]*Te[t][d]   (4 partials per column)
    {
        int d0 = tid & 63, ch = tid >> 6;
        float acc = 0.0f;
#pragma unroll
        for (int e = 0; e < 16; ++e) {
            int t = ch * 16 + e;
            acc += Srow[t] * (float)Te[t][d0];
        }
        __syncthreads();           // Y free for reuse as partial buffer
        Y[ch][d0] = acc;
        __syncthreads();
        if (tid < DD)
            out[(size_t)b * DD + tid] =
                (Y[0][tid] + Y[1][tid] + Y[2][tid] + Y[3][tid]) * invsum;
    }
}

// ---------------------------------------------------------------------------
// Kernel: new_user_emb_0 = mean_t sigmoid( relu(E@W1+b1) @ W2 + b2 )
// ---------------------------------------------------------------------------
__global__ __launch_bounds__(256) void u0_kernel(
    const int* __restrict__ h_idx, const float* __restrict__ entity,
    const _Float16* __restrict__ w1p, const float* __restrict__ b1,
    const _Float16* __restrict__ w2p, const float* __restrict__ b2,
    float* __restrict__ out) {
    __shared__ _Float16 X[TT][DD];
    __shared__ float    Y[TT][DD];
    __shared__ _Float16 X2[TT][DD];

    int b = blockIdx.x, tid = threadIdx.x;
    int wave = tid >> 5, lane = tid & 31;

    {   // gather
        int row = tid >> 2, q = tid & 3;
        const float* hp = entity + (size_t)h_idx[b * TT + row] * DD;
#pragma unroll
        for (int e = 0; e < 16; ++e) {
            int c = q * 16 + e;
            X[row][c] = (_Float16)hp[c];
        }
    }
    __syncthreads();

    gemm_block<2, 2, 4, 4, 8>(&X[0][0], DD, w1p, &Y[0][0], DD, b1, wave, lane);
    __syncthreads();
    for (int i = tid; i < TT * DD; i += 256) (&X2[0][0])[i] = (_Float16)(&Y[0][0])[i];
    __syncthreads();
    gemm_block<2, 0, 4, 4, 8>(&X2[0][0], DD, w2p, &Y[0][0], DD, b2, wave, lane);
    __syncthreads();

    if (tid < DD) {
        float acc = 0.0f;
#pragma unroll
        for (int t = 0; t < TT; ++t) acc += sigmoidf_(Y[t][tid]);
        out[(size_t)b * DD + tid] = acc * (1.0f / (float)TT);
    }
}

// ---------------------------------------------------------------------------
// Kernel: out[b] = entity[items[b]]   (B x 64 gather)
// ---------------------------------------------------------------------------
__global__ void gather_kernel(const int* __restrict__ items,
                              const float* __restrict__ entity,
                              float* __restrict__ out) {
    int i = blockIdx.x * blockDim.x + threadIdx.x;   // over B*D
    int b = i >> 6, d = i & 63;
    out[i] = entity[(size_t)items[b] * DD + d];
}

// ---------------------------------------------------------------------------
// Kernel: out[b] = mean_t entity[idx[b,t]]   (one block per b)
// ---------------------------------------------------------------------------
__global__ __launch_bounds__(256) void hmean_kernel(const int* __restrict__ idx,
                                                    const float* __restrict__ entity,
                                                    float* __restrict__ out) {
    __shared__ float P[4][DD];
    int b = blockIdx.x, tid = threadIdx.x;
    int d0 = tid & 63, ch = tid >> 6;
    const int* ip = idx + (size_t)b * TT;
    float acc = 0.0f;
#pragma unroll
    for (int e = 0; e < 16; ++e)
        acc += entity[(size_t)ip[ch * 16 + e] * DD + d0];
    P[ch][d0] = acc;
    __syncthreads();
    if (tid < DD)
        out[(size_t)b * DD + tid] =
            (P[0][tid] + P[1][tid] + P[2][tid] + P[3][tid]) * (1.0f / (float)TT);
}

// ---------------------------------------------------------------------------
// Kernel: o_l_attention + dig for 32 batch rows per block.
//   x = [emb, last, last*emb, origin, origin*emb]  (32 x 320)
//   att = sigmoid( lrelu(lrelu(x@ea1+b1)@ea2+b2) @ ea3 + b3 )
//   d   = lrelu( lrelu( (att*emb)@dw1 + db1 ) @ dw2 + db2 )
// ---------------------------------------------------------------------------
__global__ __launch_bounds__(256) void oldig_kernel(
    const float* __restrict__ emb, const float* __restrict__ origin,
    const float* __restrict__ last,
    const _Float16* __restrict__ ea1, const float* __restrict__ eb1,
    const _Float16* __restrict__ ea2, const float* __restrict__ eb2,
    const _Float16* __restrict__ ea3, const float* __restrict__ eb3,
    const _Float16* __restrict__ dw1, const float* __restrict__ db1,
    const _Float16* __restrict__ dw2, const float* __restrict__ db2,
    float* __restrict__ dst) {
    const int M = 32;
    __shared__ _Float16 X[M][5 * DD];
    __shared__ float    Y[M][DD];
    __shared__ _Float16 Xh[M][DD];
    __shared__ _Float16 Xh2[M][DD];
    __shared__ float    Att[M];

    int tid = threadIdx.x;
    int wave = tid >> 5, lane = tid & 31;
    int rbase = blockIdx.x * M;

    if (tid == 0) __builtin_prefetch(ea1, 0, 1);

    for (int i = tid; i < M * 5 * DD; i += 256) {
        int row = i / (5 * DD), c = i % (5 * DD);
        int seg = c >> 6, d = c & 63;
        size_t g = (size_t)(rbase + row) * DD + d;
        float e = emb[g], l = last[g], o = origin[g];
        float v = (seg == 0) ? e : (seg == 1) ? l : (seg == 2) ? l * e
                : (seg == 3) ? o : o * e;
        X[row][c] = (_Float16)v;
    }
    __syncthreads();

    // y1 = lrelu(x @ ea1 + b1) : M=32,N=64,K=320
    gemm_block<10, 1, 2, 4, 8>(&X[0][0], 5 * DD, ea1, &Y[0][0], DD, eb1, wave, lane);
    __syncthreads();
    for (int i = tid; i < M * DD; i += 256) (&Xh[0][0])[i] = (_Float16)(&Y[0][0])[i];
    __syncthreads();
    // y2 = lrelu(y1 @ ea2 + b2)
    gemm_block<2, 1, 2, 4, 8>(&Xh[0][0], DD, ea2, &Y[0][0], DD, eb2, wave, lane);
    __syncthreads();
    // att = sigmoid(y2 @ ea3 + b3)  (ea3 fetched as 8x b128)
    if (tid < M) Att[tid] = sigmoidf_(rowdot64(&Y[tid][0], ea3) + eb3[0]);
    __syncthreads();
    // att_i = att * emb  ->  Xh2
    for (int i = tid; i < M * DD; i += 256) {
        int row = i >> 6, d = i & 63;
        (&Xh2[0][0])[i] =
            (_Float16)(Att[row] * emb[(size_t)(rbase + row) * DD + d]);
    }
    __syncthreads();
    // dig: two chained 64x64 GEMMs with lrelu
    gemm_block<2, 1, 2, 4, 8>(&Xh2[0][0], DD, dw1, &Y[0][0], DD, db1, wave, lane);
    __syncthreads();
    for (int i = tid; i < M * DD; i += 256) (&Xh[0][0])[i] = (_Float16)(&Y[0][0])[i];
    __syncthreads();
    gemm_block<2, 1, 2, 4, 8>(&Xh[0][0], DD, dw2, &Y[0][0], DD, db2, wave, lane);
    __syncthreads();

    for (int i = tid; i < M * DD; i += 256) {
        int row = i >> 6, d = i & 63;
        dst[(size_t)(rbase + row) * DD + d] = Y[row][d];
    }
}

// ---------------------------------------------------------------------------
// Kernel: out[b] = sigmoid( dot(u0 + 2du0 + 2du1, i0 + 2di0 + 2di1 + hmean) )
// ---------------------------------------------------------------------------
__global__ void final_kernel(const float* __restrict__ u0,
                             const float* __restrict__ du0,
                             const float* __restrict__ du1,
                             const float* __restrict__ i0,
                             const float* __restrict__ di0,
                             const float* __restrict__ di1,
                             const float* __restrict__ hm,
                             float* __restrict__ out) {
    int b = blockIdx.x * blockDim.x + threadIdx.x;
    if (b >= BB) return;
    size_t base = (size_t)b * DD;
    float acc = 0.0f;
#pragma unroll 8
    for (int d = 0; d < DD; ++d) {
        float eu = u0[base + d] + 2.0f * (du0[base + d] + du1[base + d]);
        float ev = i0[base + d] + 2.0f * (di0[base + d] + di1[base + d]) + hm[base + d];
        acc += eu * ev;
    }
    out[b] = sigmoidf_(acc);
}

// ---------------------------------------------------------------------------
// Host launcher
// ---------------------------------------------------------------------------
extern "C" void kernel_launch(void* const* d_in, const int* in_sizes, int n_in,
                              void* d_out, int out_size, void* d_ws, size_t ws_size,
                              hipStream_t stream) {
    (void)in_sizes; (void)n_in; (void)out_size; (void)ws_size;

    const int*   items     = (const int*)d_in[0];
    const int*   user_h    = (const int*)d_in[1];
    const int*   user_r    = (const int*)d_in[2];
    const int*   user_t    = (const int*)d_in[3];
    const int*   item_h    = (const int*)d_in[4];
    const int*   item_r    = (const int*)d_in[5];
    const int*   item_t    = (const int*)d_in[6];
    const float* entity    = (const float*)d_in[7];
    const float* relation  = (const float*)d_in[8];
    const float* ta_w1 = (const float*)d_in[9];  const float* ta_b1 = (const float*)d_in[10];
    const float* ta_w2 = (const float*)d_in[11]; const float* ta_b2 = (const float*)d_in[12];
    const float* ea_w1 = (const float*)d_in[13]; const float* ea_b1 = (const float*)d_in[14];
    const float* ea_w2 = (const float*)d_in[15]; const float* ea_b2 = (const float*)d_in[16];
    const float* ea_w3 = (const float*)d_in[17]; const float* ea_b3 = (const float*)d_in[18];
    const float* u0_w1 = (const float*)d_in[19]; const float* u0_b1 = (const float*)d_in[20];
    const float* u0_w2 = (const float*)d_in[21]; const float* u0_b2 = (const float*)d_in[22];
    const float* ud_w1 = (const float*)d_in[23]; const float* ud_b1 = (const float*)d_in[24];
    const float* ud_w2 = (const float*)d_in[25]; const float* ud_b2 = (const float*)d_in[26];
    const float* id_w1 = (const float*)d_in[27]; const float* id_b1 = (const float*)d_in[28];
    const float* id_w2 = (const float*)d_in[29]; const float* id_b2 = (const float*)d_in[30];

    // ---- workspace carve-up ----
    char* base = (char*)d_ws;
    size_t off = 0;
    auto carve = [&](size_t bytes) -> char* {
        char* p = base + off;
        off += (bytes + 255) & ~(size_t)255;
        return p;
    };
    _Float16* hw_ta1 = (_Float16*)carve(128 * 64 * 2);   // packed
    _Float16* hw_ta2 = (_Float16*)carve(64 * 2);         // plain vector
    _Float16* hw_ea1 = (_Float16*)carve(320 * 64 * 2);   // packed
    _Float16* hw_ea2 = (_Float16*)carve(64 * 64 * 2);    // packed
    _Float16* hw_ea3 = (_Float16*)carve(64 * 2);         // plain vector
    _Float16* hw_u01 = (_Float16*)carve(64 * 64 * 2);    // packed
    _Float16* hw_u02 = (_Float16*)carve(64 * 64 * 2);    // packed
    _Float16* hw_ud1 = (_Float16*)carve(64 * 64 * 2);    // packed
    _Float16* hw_ud2 = (_Float16*)carve(64 * 64 * 2);    // packed
    _Float16* hw_id1 = (_Float16*)carve(64 * 64 * 2);    // packed
    _Float16* hw_id2 = (_Float16*)carve(64 * 64 * 2);    // packed
    size_t fb = (size_t)BB * DD * sizeof(float);
    float* f_user0 = (float*)carve(fb);
    float* f_item0 = (float*)carve(fb);
    float* f_hmean = (float*)carve(fb);
    float* f_embu0 = (float*)carve(fb);
    float* f_embu1 = (float*)carve(fb);
    float* f_embv0 = (float*)carve(fb);
    float* f_embv1 = (float*)carve(fb);
    float* f_du0   = (float*)carve(fb);
    float* f_du1   = (float*)carve(fb);
    float* f_di0   = (float*)carve(fb);
    float* f_di1   = (float*)carve(fb);

    // ---- 1. weight prep: pack matrices into B-fragment layout, cvt vectors ----
    auto pack = [&](const float* s, _Float16* d, int K, int N) {
        int n = K * N;
        pack_b_kernel<<<(n + 255) / 256, 256, 0, stream>>>(s, d, K, N);
    };
    auto cvt = [&](const float* s, _Float16* d, int n) {
        cvt_f16_kernel<<<(n + 255) / 256, 256, 0, stream>>>(s, d, n);
    };
    pack(ta_w1, hw_ta1, 128, 64);
    cvt (ta_w2, hw_ta2, 64);
    pack(ea_w1, hw_ea1, 320, 64);
    pack(ea_w2, hw_ea2, 64, 64);
    cvt (ea_w3, hw_ea3, 64);
    pack(u0_w1, hw_u01, 64, 64);
    pack(u0_w2, hw_u02, 64, 64);
    pack(ud_w1, hw_ud1, 64, 64);
    pack(ud_w2, hw_ud2, 64, 64);
    pack(id_w1, hw_id1, 64, 64);
    pack(id_w2, hw_id2, 64, 64);

    const size_t LS = (size_t)BB * TT;  // layer stride in index arrays

    // ---- 2. knowledge attention (4 independent launches) ----
    ka_kernel<<<BB, 256, 0, stream>>>(user_h,      user_r,      user_t,      entity, relation,
                                      hw_ta1, ta_b1, hw_ta2, ta_b2, f_embu0);
    ka_kernel<<<BB, 256, 0, stream>>>(user_h + LS, user_r + LS, user_t + LS, entity, relation,
                                      hw_ta1, ta_b1, hw_ta2, ta_b2, f_embu1);
    ka_kernel<<<BB, 256, 0, stream>>>(item_h,      item_r,      item_t,      entity, relation,
                                      hw_ta1, ta_b1, hw_ta2, ta_b2, f_embv0);
    ka_kernel<<<BB, 256, 0, stream>>>(item_h + LS, item_r + LS, item_t + LS, entity, relation,
                                      hw_ta1, ta_b1, hw_ta2, ta_b2, f_embv1);

    // ---- 3. user_0 path / item seeds ----
    u0_kernel<<<BB, 256, 0, stream>>>(user_h, entity, hw_u01, u0_b1, hw_u02, u0_b2, f_user0);
    gather_kernel<<<(BB * DD) / 256, 256, 0, stream>>>(items, entity, f_item0);
    hmean_kernel<<<BB, 256, 0, stream>>>(item_h, entity, f_hmean);

    // ---- 4. o_l_attention + dig (ordered: layer 1 depends on layer 0) ----
    const int OG = BB / 32;
    oldig_kernel<<<OG, 256, 0, stream>>>(f_embu0, f_user0, f_user0,
                                         hw_ea1, ea_b1, hw_ea2, ea_b2, hw_ea3, ea_b3,
                                         hw_ud1, ud_b1, hw_ud2, ud_b2, f_du0);
    oldig_kernel<<<OG, 256, 0, stream>>>(f_embu1, f_user0, f_du0,
                                         hw_ea1, ea_b1, hw_ea2, ea_b2, hw_ea3, ea_b3,
                                         hw_ud1, ud_b1, hw_ud2, ud_b2, f_du1);
    oldig_kernel<<<OG, 256, 0, stream>>>(f_embv0, f_item0, f_item0,
                                         hw_ea1, ea_b1, hw_ea2, ea_b2, hw_ea3, ea_b3,
                                         hw_id1, id_b1, hw_id2, id_b2, f_di0);
    oldig_kernel<<<OG, 256, 0, stream>>>(f_embv1, f_item0, f_di0,
                                         hw_ea1, ea_b1, hw_ea2, ea_b2, hw_ea3, ea_b3,
                                         hw_id1, id_b1, hw_id2, id_b2, f_di1);

    // ---- 5. final score ----
    final_kernel<<<BB / 256, 256, 0, stream>>>(f_user0, f_du0, f_du1,
                                               f_item0, f_di0, f_di1, f_hmean,
                                               (float*)d_out);
}
